// RN_79542794322129
// MI455X (gfx1250) — matrix operand order repaired and checked by reading, not compiled, not run on hardware
//
#include <hip/hip_runtime.h>

// ---------------------------------------------------------------------------
// CDNA5 (gfx1250) wave32 WMMA implementation.
// Shapes: BS=64, NR=36, VDIM=2048(+4 pos), QDIM=1024, CSUB=512, RDIM=256
// ---------------------------------------------------------------------------

typedef __attribute__((ext_vector_type(16))) __bf16 bf16x16;
typedef __attribute__((ext_vector_type(8)))  float  f32x8;
typedef __attribute__((ext_vector_type(4)))  int    i32x4;

union FragU { bf16x16 v; i32x4 q[2]; };

// A-fragment (16x32 bf16, M x K), ISA 7.12.2 layout:
// lane m = lane&15, half h = lane>>4; element runs k=[8h..8h+7] and [16+8h..+7]
__device__ __forceinline__ bf16x16 ldA(const __bf16* base, int ld, int row0,
                                       int kbase, int lane) {
  int m = lane & 15, hh = lane >> 4;
  const __bf16* p = base + (size_t)(row0 + m) * ld + (kbase + 8 * hh);
  FragU f;
  f.q[0] = *(const i32x4*)p;
  f.q[1] = *(const i32x4*)(p + 16);
  return f.v;
}

// B-fragment (32x16 bf16, K x N) from an N-major (transposed) weight matrix:
// lane column n = lane&15, half h = lane>>4; element runs k=[8h..8h+7] and [16+8h..+7]
__device__ __forceinline__ bf16x16 ldB(const __bf16* bt, int ldk, int n0,
                                       int kbase, int lane) {
  int n = lane & 15, hh = lane >> 4;
  const __bf16* p = bt + (size_t)(n0 + n) * ldk + (kbase + 8 * hh);
  FragU f;
  f.q[0] = *(const i32x4*)p;
  f.q[1] = *(const i32x4*)(p + 16);
  return f.v;
}

#define WMMA_BF16(a, b, c) \
  __builtin_amdgcn_wmma_f32_16x16x32_bf16(false, (a), false, (b), (short)0, (c), false, false)

// ------------------------- conversion kernels ------------------------------

__global__ void k_cvt(const float* __restrict__ in, __bf16* __restrict__ out, int n) {
  for (int i = blockIdx.x * blockDim.x + threadIdx.x; i < n; i += gridDim.x * blockDim.x)
    out[i] = (__bf16)in[i];
}

// in: [K][N] f32 row-major -> out: [N][Kout] bf16 (N-major), k < Kout <= K
__global__ void k_cvtT(const float* __restrict__ in, __bf16* __restrict__ out,
                       int N, int Kout, int total) {
  for (int idx = blockIdx.x * blockDim.x + threadIdx.x; idx < total;
       idx += gridDim.x * blockDim.x) {
    int n = idx / Kout;
    int k = idx - n * Kout;
    out[(size_t)n * Kout + k] = (__bf16)in[(size_t)k * N + n];
  }
}

// ------------------------- Qp = relu(Q @ Wq + bq) --------------------------
// M=64, N=512, K=1024. grid=32 (one 16-col tile per block), block=128 (4 waves,
// one 16-row tile per wave).
__global__ __launch_bounds__(128) void k_qp(const __bf16* __restrict__ Qb,
                                            const __bf16* __restrict__ Wqt,
                                            const float* __restrict__ bq,
                                            float* __restrict__ Qp) {
  int lane = threadIdx.x & 31, wave = threadIdx.x >> 5;
  int M0 = wave * 16, N0 = blockIdx.x * 16;
  f32x8 acc = {};
  for (int kk = 0; kk < 32; ++kk) {
    int kb = kk * 32;
    bf16x16 a = ldA(Qb, 1024, M0, kb, lane);
    bf16x16 b = ldB(Wqt, 1024, N0, kb, lane);
    acc = WMMA_BF16(a, b, acc);
  }
  int n = lane & 15, hh = lane >> 4, col = N0 + n;
  float bias = bq[col];
#pragma unroll
  for (int v = 0; v < 8; ++v) {
    int row = M0 + 8 * hh + v;
    float x = fmaxf(acc[v] + bias, 0.f);
    Qp[(size_t)row * 512 + col] = x;
  }
}

// ---- Xs = relu(X@Wv[:2048] + pos@Wv[2048:] + bv) + Qp[b] ------------------
// M=2304, N=512, K=2048. grid=36*8, block=256 (8 waves). Each wave: 1 M-tile,
// 2 N-tiles.
__global__ __launch_bounds__(256) void k_xs(const __bf16* __restrict__ Xb,
                                            const __bf16* __restrict__ Wvt,
                                            const float* __restrict__ Wv,
                                            const float* __restrict__ bv,
                                            const float* __restrict__ pos,
                                            const float* __restrict__ Qp,
                                            float* __restrict__ Xs) {
  int lane = threadIdx.x & 31, wave = threadIdx.x >> 5;
  int mblk = blockIdx.x % 36, nblk = blockIdx.x / 36;
  int M0 = mblk * 64 + (wave & 3) * 16;
  int Nb = nblk * 64;
  int nt0 = (wave >> 2) * 2;
  f32x8 acc[2] = {};
  for (int kk = 0; kk < 64; ++kk) {
    int kb = kk * 32;
    bf16x16 a = ldA(Xb, 2048, M0, kb, lane);
#pragma unroll
    for (int j = 0; j < 2; ++j) {
      bf16x16 b = ldB(Wvt, 2048, Nb + (nt0 + j) * 16, kb, lane);
      acc[j] = WMMA_BF16(a, b, acc[j]);
    }
  }
  int n = lane & 15, hh = lane >> 4;
#pragma unroll
  for (int j = 0; j < 2; ++j) {
    int col = Nb + (nt0 + j) * 16 + n;
    float bias = bv[col];
    float w0 = Wv[(size_t)2048 * 512 + col];
    float w1 = Wv[(size_t)2049 * 512 + col];
    float w2 = Wv[(size_t)2050 * 512 + col];
    float w3 = Wv[(size_t)2051 * 512 + col];
#pragma unroll
    for (int v = 0; v < 8; ++v) {
      int grow = M0 + 8 * hh + v;                 // grow = b*36 + r
      const float* pp = pos + (size_t)grow * 4;
      float x = acc[j][v] + bias + pp[0] * w0 + pp[1] * w1 + pp[2] * w2 + pp[3] * w3;
      x = fmaxf(x, 0.f) + Qp[(size_t)(grow / 36) * 512 + col];
      Xs[(size_t)grow * 512 + col] = x;
    }
  }
}

// ---- fused: pair -> relu(pair@W1+b1) -> relu(h@W2+b2) ---------------------
// One block per (b, i). 8 waves; wave w owns output columns [32w, 32w+32).
// pair rows = j (padded 36->48). K=512 staged in two 256-wide LDS chunks.
__global__ __launch_bounds__(256) void k_rel(const float* __restrict__ Xs,
                                             const __bf16* __restrict__ W1t,
                                             const float* __restrict__ b1,
                                             const __bf16* __restrict__ W2t,
                                             const float* __restrict__ b2,
                                             float* __restrict__ out) {
  __shared__ __align__(16) __bf16 pair_s[48 * 256];
  __shared__ __align__(16) __bf16 h_s[48 * 256];
  __shared__ float xsi[512];

  int tid = threadIdx.x, lane = tid & 31, wave = tid >> 5;
  int b = blockIdx.x / 36, i = blockIdx.x % 36;
  const float* Xsb = Xs + (size_t)b * 36 * 512;

  xsi[tid]       = Xsb[(size_t)i * 512 + tid];
  xsi[tid + 256] = Xsb[(size_t)i * 512 + tid + 256];
  __syncthreads();

  int N0 = wave * 32;
  f32x8 acc[3][2] = {};

  for (int chunk = 0; chunk < 2; ++chunk) {
    int koff = chunk * 256;
    // build bf16 pair chunk: pair[j][k] = Xs[b,j,k] * Xs[b,i,k]
    for (int m = 0; m < 48; ++m) {
      float val = 0.f;
      if (m < 36) val = Xsb[(size_t)m * 512 + koff + tid] * xsi[koff + tid];
      pair_s[m * 256 + tid] = (__bf16)val;
    }
    __syncthreads();
    for (int kk = 0; kk < 8; ++kk) {
      int kb = kk * 32;
      bf16x16 a0 = ldA(pair_s, 256, 0,  kb, lane);
      bf16x16 a1 = ldA(pair_s, 256, 16, kb, lane);
      bf16x16 a2 = ldA(pair_s, 256, 32, kb, lane);
#pragma unroll
      for (int j = 0; j < 2; ++j) {
        bf16x16 bb = ldB(W1t, 512, N0 + j * 16, koff + kb, lane);
        acc[0][j] = WMMA_BF16(a0, bb, acc[0][j]);
        acc[1][j] = WMMA_BF16(a1, bb, acc[1][j]);
        acc[2][j] = WMMA_BF16(a2, bb, acc[2][j]);
      }
    }
    __syncthreads();
  }

  // h = relu(acc + b1) -> LDS bf16 (full 48x256 tile, restriped as A for GEMM2)
  int n = lane & 15, hh = lane >> 4;
#pragma unroll
  for (int j = 0; j < 2; ++j) {
    int col = N0 + j * 16 + n;
    float bias = b1[col];
#pragma unroll
    for (int mt = 0; mt < 3; ++mt)
#pragma unroll
      for (int v = 0; v < 8; ++v) {
        int m = mt * 16 + hh * 8 + v;
        float x = fmaxf(acc[mt][j][v] + bias, 0.f);
        h_s[m * 256 + col] = (__bf16)x;
      }
  }
  __syncthreads();

  // GEMM2: r = relu(h @ W2 + b2)
  f32x8 acc2[3][2] = {};
  for (int kk = 0; kk < 8; ++kk) {
    int kb = kk * 32;
    bf16x16 a0 = ldA(h_s, 256, 0,  kb, lane);
    bf16x16 a1 = ldA(h_s, 256, 16, kb, lane);
    bf16x16 a2 = ldA(h_s, 256, 32, kb, lane);
#pragma unroll
    for (int j = 0; j < 2; ++j) {
      bf16x16 bb = ldB(W2t, 256, N0 + j * 16, kb, lane);
      acc2[0][j] = WMMA_BF16(a0, bb, acc2[0][j]);
      acc2[1][j] = WMMA_BF16(a1, bb, acc2[1][j]);
      acc2[2][j] = WMMA_BF16(a2, bb, acc2[2][j]);
    }
  }

  float* ob = out + (size_t)(b * 36 + i) * 36 * 256;
#pragma unroll
  for (int j = 0; j < 2; ++j) {
    int col = N0 + j * 16 + n;
    float bias = b2[col];
#pragma unroll
    for (int mt = 0; mt < 3; ++mt)
#pragma unroll
      for (int v = 0; v < 8; ++v) {
        int m = mt * 16 + hh * 8 + v;
        if (m < 36) {
          float x = fmaxf(acc2[mt][j][v] + bias, 0.f);
          ob[(size_t)m * 256 + col] = x;
        }
      }
  }
}

// ---------------------------------------------------------------------------

extern "C" void kernel_launch(void* const* d_in, const int* in_sizes, int n_in,
                              void* d_out, int out_size, void* d_ws, size_t ws_size,
                              hipStream_t stream) {
  (void)in_sizes; (void)n_in; (void)out_size; (void)ws_size;
  const float* X   = (const float*)d_in[0];   // [64,36,2048]
  const float* Q   = (const float*)d_in[1];   // [64,1024]
  const float* pos = (const float*)d_in[2];   // [64,36,4]
  const float* Wv  = (const float*)d_in[3];   // [2052,512]
  const float* bv  = (const float*)d_in[4];   // [512]
  const float* Wq  = (const float*)d_in[5];   // [1024,512]
  const float* bq  = (const float*)d_in[6];   // [512]
  const float* W1  = (const float*)d_in[7];   // [512,256]
  const float* b1  = (const float*)d_in[8];   // [256]
  const float* W2  = (const float*)d_in[9];   // [256,256]
  const float* b2  = (const float*)d_in[10];  // [256]
  float* out = (float*)d_out;                 // [64,36,36,256]

  char* ws = (char*)d_ws;
  size_t off = 0;
  auto alloc = [&](size_t bytes) -> void* {
    void* p = ws + off;
    off = (off + bytes + 255) & ~((size_t)255);
    return p;
  };
  __bf16* Xb  = (__bf16*)alloc((size_t)2304 * 2048 * 2);  // X in bf16
  __bf16* Qb  = (__bf16*)alloc((size_t)64 * 1024 * 2);    // Q in bf16
  __bf16* Wvt = (__bf16*)alloc((size_t)512 * 2048 * 2);   // Wv[:2048]^T bf16
  __bf16* Wqt = (__bf16*)alloc((size_t)512 * 1024 * 2);   // Wq^T bf16
  __bf16* W1t = (__bf16*)alloc((size_t)256 * 512 * 2);    // W1^T bf16
  __bf16* W2t = (__bf16*)alloc((size_t)256 * 256 * 2);    // W2^T bf16
  float*  Qp  = (float*)alloc((size_t)64 * 512 * 4);      // relu(Q@Wq+bq)
  float*  Xs  = (float*)alloc((size_t)2304 * 512 * 4);    // projected+broadcast

  // precision conversion / weight transposition (bandwidth-trivial)
  k_cvt<<<4096, 256, 0, stream>>>(X, Xb, 2304 * 2048);
  k_cvt<<<256, 256, 0, stream>>>(Q, Qb, 64 * 1024);
  k_cvtT<<<2048, 256, 0, stream>>>(Wv, Wvt, 512, 2048, 512 * 2048);
  k_cvtT<<<1024, 256, 0, stream>>>(Wq, Wqt, 512, 1024, 512 * 1024);
  k_cvtT<<<512, 256, 0, stream>>>(W1, W1t, 256, 512, 256 * 512);
  k_cvtT<<<256, 256, 0, stream>>>(W2, W2t, 256, 256, 256 * 256);

  // projections
  k_qp<<<32, 128, 0, stream>>>(Qb, Wqt, bq, Qp);
  k_xs<<<36 * 8, 256, 0, stream>>>(Xb, Wvt, Wv, bv, pos, Qp, Xs);

  // fused pair -> conv1 -> conv2 (dominant compute, LDS-fused, L2-resident Xs)
  k_rel<<<64 * 36, 256, 0, stream>>>(Xs, W1t, b1, W2t, b2, out);
}
